// MultiHeadLatentAttention_24223615550285
// MI455X (gfx1250) — matrix operand order repaired and checked by reading, not compiled, not run on hardware
//
#include <hip/hip_runtime.h>
#include <hip/hip_bf16.h>
#include <stdint.h>

// ---------------------------------------------------------------------------
// MLA forward on gfx1250: bf16 WMMA (v_wmma_f32_16x16x32_bf16) everywhere.
// ---------------------------------------------------------------------------

typedef __attribute__((ext_vector_type(16))) __bf16        v16bf;
typedef __attribute__((ext_vector_type(8)))  float         v8f;
typedef __attribute__((ext_vector_type(4)))  unsigned int  u32x4;

union FragU {
  struct { u32x4 lo, hi; } p;
  v16bf v;
};
static_assert(sizeof(v16bf) == 32, "v16bf must be 32 bytes");
static_assert(sizeof(FragU) == 32, "FragU must be 32 bytes");

__device__ __forceinline__ uint16_t f2bf(float f) {
  uint32_t u = __float_as_uint(f);
  u += 0x7FFFu + ((u >> 16) & 1u);        // round-to-nearest-even
  return (uint16_t)(u >> 16);
}

// A fragment (16x32 bf16), source row-major MxK, leading dim lda (elements).
// ISA layout: lane l holds row m=l%16; v0..3 = K (l/16)*8 + 0..7, v4..7 = +16.
__device__ __forceinline__ v16bf load_a_frag(const uint16_t* A, int lda,
                                             int m0, int k0, int lane) {
  const uint16_t* p = A + (size_t)(m0 + (lane & 15)) * lda
                        + (size_t)(k0 + ((lane >> 4) << 3));
  FragU f;
  f.p.lo = *reinterpret_cast<const u32x4*>(p);
  f.p.hi = *reinterpret_cast<const u32x4*>(p + 16);
  return f.v;
}

// B fragment (32x16 bf16, KxN), source is N-major: Bt[n][k], leading dim ldb.
// ISA layout: lane l holds col n=l%16; v0..3 = K (l/16)*8 + 0..7, v4..7 = +16.
__device__ __forceinline__ v16bf load_b_frag(const uint16_t* Bt, int ldb,
                                             int n0, int k0, int lane) {
  const uint16_t* p = Bt + (size_t)(n0 + (lane & 15)) * ldb
                         + (size_t)(k0 + ((lane >> 4) << 3));
  FragU f;
  f.p.lo = *reinterpret_cast<const u32x4*>(p);
  f.p.hi = *reinterpret_cast<const u32x4*>(p + 16);
  return f.v;
}

// A fragment of the staged P tile in LDS (16x32, lda=32). memcpy keeps the
// compiler from reordering these ds loads past the uint16 ds stores.
__device__ __forceinline__ v16bf load_p_frag(const uint16_t* Pw, int lane) {
  const uint16_t* p = Pw + (lane & 15) * 32 + ((lane >> 4) << 3);
  FragU f;
  __builtin_memcpy(&f.p.lo, p, 16);
  __builtin_memcpy(&f.p.hi, p + 16, 16);
  return f.v;
}

__device__ __forceinline__ v8f wmma_bf16(v16bf a, v16bf b, v8f c) {
  return __builtin_amdgcn_wmma_f32_16x16x32_bf16(false, a, false, b,
                                                 (short)0, c, false, false);
}

// ---------------------------------------------------------------------------
// Elementwise helpers
// ---------------------------------------------------------------------------

__global__ void cast_bf16_kernel(const float* __restrict__ src,
                                 uint16_t* __restrict__ dst, size_t n) {
  size_t i = (size_t)blockIdx.x * blockDim.x + threadIdx.x;
  if (i < n) dst[i] = f2bf(src[i]);
}

// Wt[n][k] = bf16(W[k][n]); W is row-major KxN. idx: k fastest.
__global__ void transpose_cast_kernel(const float* __restrict__ W,
                                      uint16_t* __restrict__ Wt, int K, int N) {
  size_t idx = (size_t)blockIdx.x * blockDim.x + threadIdx.x;
  if (idx >= (size_t)K * N) return;
  int    k = (int)(idx % K);
  size_t n = idx / K;
  Wt[idx] = f2bf(W[(size_t)k * N + n]);
}

// Build q/k bf16 buffers (B,H,S,256): first 128 dims copied from `lo`,
// last 128 dims = rope(hi). lo/hi are row-major (B*S, 2048) f32.
__global__ void pack_rope_kernel(const float* __restrict__ lo,
                                 const float* __restrict__ hi,
                                 uint16_t* __restrict__ dst, int S) {
  size_t idx  = (size_t)blockIdx.x * blockDim.x + threadIdx.x; // B*16*S*128
  int    pair = (int)(idx & 127);
  size_t rest = idx >> 7;                 // (b*16+h)*S + s
  int    s    = (int)(rest % S);
  size_t bh   = rest / S;
  int    h    = (int)(bh & 15);
  size_t b    = bh >> 4;
  const size_t srow = ((size_t)b * S + s) * 2048 + (size_t)h * 128;
  uint16_t* o = dst + rest * 256;
  if (pair < 64) {
    o[2 * pair]     = f2bf(lo[srow + 2 * pair]);
    o[2 * pair + 1] = f2bf(lo[srow + 2 * pair + 1]);
  } else {
    int   i  = pair - 64;
    float x1 = hi[srow + 2 * i];
    float x2 = hi[srow + 2 * i + 1];
    // inv_freq = 10000^(-2i/128)
    float fr = __expf(-((float)(2 * i) * (1.0f / 128.0f)) * 9.210340372f);
    float ang = (float)s * fr;
    float c = cosf(ang), sn = sinf(ang);
    o[128 + 2 * i]     = f2bf(x1 * c - x2 * sn);
    o[128 + 2 * i + 1] = f2bf(x1 * sn + x2 * c);
  }
}

// vT[(b*16+h)*128 + d][s] = bf16(vc[(b*S+s)][h*128+d]); idx: s fastest.
__global__ void pack_vT_kernel(const float* __restrict__ vc,
                               uint16_t* __restrict__ vT, int S) {
  size_t idx  = (size_t)blockIdx.x * blockDim.x + threadIdx.x; // B*16*128*S
  int    s    = (int)(idx % S);
  size_t rest = idx / S;
  int    d    = (int)(rest & 127);
  size_t bh   = rest >> 7;
  int    h    = (int)(bh & 15);
  size_t b    = bh >> 4;
  vT[idx] = f2bf(vc[((size_t)b * S + s) * 2048 + (size_t)h * 128 + d]);
}

// ---------------------------------------------------------------------------
// GEMM: C(MxN,f32) = A(bf16 MxK row-major) * Bt(bf16 NxK row-major)^T + bias
// 128x128 block, 8 waves (2x4), each wave 64x32 via 4x2 WMMA tiles.
// ---------------------------------------------------------------------------
__global__ __launch_bounds__(256)
void gemm_bf16_nt(const uint16_t* __restrict__ A, const uint16_t* __restrict__ Bt,
                  const float* __restrict__ bias, float* __restrict__ C,
                  int M, int N, int K) {
  const int lane = threadIdx.x & 31;
  const int wave = threadIdx.x >> 5;
  const int wm = wave & 1, wn = wave >> 1;
  const int m0 = blockIdx.x * 128 + wm * 64;
  const int n0 = blockIdx.y * 128 + wn * 32;
  (void)M;
  v8f acc[4][2] = {};
  for (int k0 = 0; k0 < K; k0 += 32) {
    v16bf b0 = load_b_frag(Bt, K, n0,      k0, lane);
    v16bf b1 = load_b_frag(Bt, K, n0 + 16, k0, lane);
#pragma unroll
    for (int i = 0; i < 4; ++i) {
      v16bf a = load_a_frag(A, K, m0 + i * 16, k0, lane);
      acc[i][0] = wmma_bf16(a, b0, acc[i][0]);
      acc[i][1] = wmma_bf16(a, b1, acc[i][1]);
    }
  }
  const int nl = lane & 15, half = lane >> 4;
  const float bias0 = bias[n0 + nl];
  const float bias1 = bias[n0 + 16 + nl];
#pragma unroll
  for (int i = 0; i < 4; ++i) {
    const int mb = m0 + i * 16 + 8 * half;
#pragma unroll
    for (int r = 0; r < 8; ++r) {
      float* row = C + (size_t)(mb + r) * N + n0;
      row[nl]      = acc[i][0][r] + bias0;
      row[16 + nl] = acc[i][1][r] + bias1;
    }
  }
}

// ---------------------------------------------------------------------------
// Flash-style causal attention. grid = (S/128, H, B), 256 threads (8 waves).
// Wave w owns 16 q rows; q frags (K=256) and 16x128 ctx acc stay in VGPRs.
// ctx written as f32 row-major (B*S, H*128) ready for the output GEMM.
// ---------------------------------------------------------------------------
__global__ __launch_bounds__(256)
void mla_attn_kernel(const uint16_t* __restrict__ qB, const uint16_t* __restrict__ kB,
                     const uint16_t* __restrict__ vT, float* __restrict__ ctx, int S) {
  __shared__ __align__(16) uint16_t Pst[8][16 * 32];
  const int lane = threadIdx.x & 31;
  const int wave = threadIdx.x >> 5;
  const int half = lane >> 4, nl = lane & 15;
  const int h = blockIdx.y, b = blockIdx.z;
  const int qrow = blockIdx.x * 128 + wave * 16;
  const size_t bh = (size_t)(b * 16 + h);
  const uint16_t* q = qB + bh * (size_t)S * 256;
  const uint16_t* k = kB + bh * (size_t)S * 256;
  const uint16_t* v = vT + bh * (size_t)128 * S;

  v16bf qf[8];
#pragma unroll
  for (int t = 0; t < 8; ++t) qf[t] = load_a_frag(q, 256, qrow, t * 32, lane);

  v8f cacc[8] = {};
  float m8[8], l8[8];
#pragma unroll
  for (int r = 0; r < 8; ++r) { m8[r] = -1e30f; l8[r] = 0.0f; }

  const float scale = 0.0625f; // 1/sqrt(256)
  const int jend = qrow + 15;
  for (int j0 = 0; j0 <= jend; j0 += 32) {
    v8f s0 = {}, s1 = {};
#pragma unroll
    for (int t = 0; t < 8; ++t) {
      v16bf k0f = load_b_frag(k, 256, j0,      t * 32, lane);
      v16bf k1f = load_b_frag(k, 256, j0 + 16, t * 32, lane);
      s0 = wmma_bf16(qf[t], k0f, s0);
      s1 = wmma_bf16(qf[t], k1f, s1);
    }
    const int col0 = j0 + nl, col1 = col0 + 16;
    const int rbase = qrow + 8 * half;
    float p0[8], p1[8];
#pragma unroll
    for (int r = 0; r < 8; ++r) {
      const int qi = rbase + r;
      float e0 = (col0 <= qi) ? s0[r] * scale : -1e30f;
      float e1 = (col1 <= qi) ? s1[r] * scale : -1e30f;
      float rm = fmaxf(e0, e1);
      rm = fmaxf(rm, __shfl_xor(rm, 1, 32));
      rm = fmaxf(rm, __shfl_xor(rm, 2, 32));
      rm = fmaxf(rm, __shfl_xor(rm, 4, 32));
      rm = fmaxf(rm, __shfl_xor(rm, 8, 32));
      float mn = fmaxf(m8[r], rm);
      float alpha = __expf(m8[r] - mn);
      e0 = __expf(e0 - mn);
      e1 = __expf(e1 - mn);
      float rs = e0 + e1;
      rs += __shfl_xor(rs, 1, 32);
      rs += __shfl_xor(rs, 2, 32);
      rs += __shfl_xor(rs, 4, 32);
      rs += __shfl_xor(rs, 8, 32);
      l8[r] = l8[r] * alpha + rs;
      m8[r] = mn;
      p0[r] = e0;
      p1[r] = e1;
#pragma unroll
      for (int t = 0; t < 8; ++t) cacc[t][r] = cacc[t][r] * alpha;
    }
    // Stage P (C-layout -> 16x32 bf16 tile), re-read in A-fragment layout.
#pragma unroll
    for (int r = 0; r < 8; ++r) {
      const int m = r + 8 * half;
      Pst[wave][m * 32 + nl]      = f2bf(p0[r]);
      Pst[wave][m * 32 + 16 + nl] = f2bf(p1[r]);
    }
    v16bf pf = load_p_frag(&Pst[wave][0], lane);
#pragma unroll
    for (int t = 0; t < 8; ++t) {
      v16bf vf = load_b_frag(v, S, t * 16, j0, lane);
      cacc[t] = wmma_bf16(pf, vf, cacc[t]);
    }
  }
#pragma unroll
  for (int r = 0; r < 8; ++r) {
    const float inv = 1.0f / l8[r];
    const int qi = qrow + 8 * half + r;
    float* crow = ctx + ((size_t)(b * S + qi)) * 2048 + (size_t)h * 128;
#pragma unroll
    for (int t = 0; t < 8; ++t) crow[t * 16 + nl] = cacc[t][r] * inv;
  }
}

// ---------------------------------------------------------------------------
// Host-side orchestration
// ---------------------------------------------------------------------------
extern "C" void kernel_launch(void* const* d_in, const int* in_sizes, int n_in,
                              void* d_out, int out_size, void* d_ws, size_t ws_size,
                              hipStream_t stream) {
  (void)in_sizes; (void)n_in; (void)out_size; (void)ws_size;
  const float* h_in = (const float*)d_in[0];
  const float* WDKV = (const float*)d_in[1];
  const float* bDKV = (const float*)d_in[2];
  const float* WUK  = (const float*)d_in[3];
  const float* bUK  = (const float*)d_in[4];
  const float* WUV  = (const float*)d_in[5];
  const float* bUV  = (const float*)d_in[6];
  const float* WDQ  = (const float*)d_in[7];
  const float* bDQ  = (const float*)d_in[8];
  const float* WUQ  = (const float*)d_in[9];
  const float* bUQ  = (const float*)d_in[10];
  const float* WKR  = (const float*)d_in[11];
  const float* bKR  = (const float*)d_in[12];
  const float* WO   = (const float*)d_in[13];
  const float* bO   = (const float*)d_in[14];

  constexpr int Bb = 2, S = 2048, H = 16, DM = 2048, DKV = 512, DQ = 1536;
  constexpr int M = Bb * S; // 4096

  float* out  = (float*)d_out;
  float* oCKV = out + (size_t)M * DM;          // (B,S,512)
  float* oKR  = oCKV + (size_t)M * DKV;        // (B,S,2048)

  char* wsp = (char*)d_ws;
  auto alloc = [&](size_t bytes) -> char* {
    char* p = wsp;
    wsp += (bytes + 255) & ~(size_t)255;
    return p;
  };
  uint16_t* hB    = (uint16_t*)alloc((size_t)M * DM * 2);
  uint16_t* WtDKV = (uint16_t*)alloc((size_t)DM * DKV * 2);
  uint16_t* WtUK  = (uint16_t*)alloc((size_t)DKV * DM * 2);
  uint16_t* WtUV  = (uint16_t*)alloc((size_t)DKV * DM * 2);
  uint16_t* WtDQ  = (uint16_t*)alloc((size_t)DM * DQ * 2);
  uint16_t* WtUQ  = (uint16_t*)alloc((size_t)DQ * DM * 2);
  uint16_t* WtKR  = (uint16_t*)alloc((size_t)DM * DM * 2);
  uint16_t* WtO   = (uint16_t*)alloc((size_t)DM * DM * 2);
  uint16_t* cKVb  = (uint16_t*)alloc((size_t)M * DKV * 2);
  float*    cQ    = (float*)   alloc((size_t)M * DQ * 4);
  uint16_t* cQb   = (uint16_t*)alloc((size_t)M * DQ * 2);
  float*    kc    = (float*)   alloc((size_t)M * DM * 4);  // reused as ctx
  float*    vc    = (float*)   alloc((size_t)M * DM * 4);
  float*    qc    = (float*)   alloc((size_t)M * DM * 4);  // reused as ctxb
  uint16_t* kBuf  = (uint16_t*)alloc((size_t)Bb * H * S * 256 * 2);
  uint16_t* qBuf  = (uint16_t*)alloc((size_t)Bb * H * S * 256 * 2);
  uint16_t* vTb   = (uint16_t*)alloc((size_t)Bb * H * 128 * S * 2);
  float*    ctx   = kc;                                    // free after pack_k
  uint16_t* ctxb  = (uint16_t*)qc;                         // free after pack_q

  auto cast = [&](const float* s, uint16_t* d, size_t n) {
    cast_bf16_kernel<<<(unsigned)((n + 255) / 256), 256, 0, stream>>>(s, d, n);
  };
  auto tcast = [&](const float* W, uint16_t* Wt, int K, int N) {
    size_t n = (size_t)K * N;
    transpose_cast_kernel<<<(unsigned)((n + 255) / 256), 256, 0, stream>>>(W, Wt, K, N);
  };
  auto gemm = [&](const uint16_t* A, const uint16_t* Bt, const float* bias,
                  float* C, int Mm, int Nn, int Kk) {
    dim3 g(Mm / 128, Nn / 128);
    gemm_bf16_nt<<<g, 256, 0, stream>>>(A, Bt, bias, C, Mm, Nn, Kk);
  };

  // Precision staging
  cast(h_in, hB, (size_t)M * DM);
  tcast(WDKV, WtDKV, DM, DKV);
  tcast(WUK,  WtUK,  DKV, DM);
  tcast(WUV,  WtUV,  DKV, DM);
  tcast(WDQ,  WtDQ,  DM, DQ);
  tcast(WUQ,  WtUQ,  DQ, DM);
  tcast(WKR,  WtKR,  DM, DM);
  tcast(WO,   WtO,   DM, DM);

  // Projections
  gemm(hB, WtDKV, bDKV, oCKV, M, DKV, DM);   // c_KV -> output segment
  cast(oCKV, cKVb, (size_t)M * DKV);
  gemm(hB, WtKR, bKR, oKR, M, DM, DM);       // kr_cache -> output segment
  gemm(hB, WtDQ, bDQ, cQ, M, DQ, DM);
  cast(cQ, cQb, (size_t)M * DQ);
  gemm(cKVb, WtUK, bUK, kc, M, DM, DKV);     // k_C (flat)
  gemm(cKVb, WtUV, bUV, vc, M, DM, DKV);     // v_C (flat)
  gemm(cQb, WtUQ, bUQ, qc, M, DM, DQ);       // q_C (flat)

  // Head split + RoPE + concat + transpose-for-WMMA
  {
    size_t n = (size_t)Bb * H * S * 128;
    pack_rope_kernel<<<(unsigned)(n / 256), 256, 0, stream>>>(kc, oKR, kBuf, S);
    pack_rope_kernel<<<(unsigned)(n / 256), 256, 0, stream>>>(qc, qc, qBuf, S);
    size_t nv = (size_t)Bb * H * 128 * S;
    pack_vT_kernel<<<(unsigned)(nv / 256), 256, 0, stream>>>(vc, vTb, S);
  }

  // Causal flash attention
  {
    dim3 g(S / 128, H, Bb);
    mla_attn_kernel<<<g, 256, 0, stream>>>(qBuf, kBuf, vTb, ctx, S);
  }

  // Output projection
  cast(ctx, ctxb, (size_t)M * DM);
  gemm(ctxb, WtO, bO, out, M, DM, DM);
}
